// AT_MLP_78048145702942
// MI455X (gfx1250) — compile-verified, matrix operand order checked by reading, not run on hardware
//
#include <hip/hip_runtime.h>

// ---------- CDNA5 WMMA types ----------
typedef __attribute__((ext_vector_type(16))) _Float16 v16h;
typedef __attribute__((ext_vector_type(8)))  _Float16 v8h;
typedef __attribute__((ext_vector_type(8)))  float    v8f;

static constexpr int BATCH = 2048;
static constexpr int NMOL  = 16;
static constexpr int HEADS = 8;
static constexpr int ADIM  = 64;
static constexpr int DIN   = 64;
static constexpr int CDIM  = 512;               // CA == CB
static constexpr int TOK   = BATCH * NMOL;      // 32768
static constexpr int KPAD  = 96;                // layer-1 K padded 70 -> 96

__device__ __forceinline__ v8f vzero8f() {
    v8f z = {0.f, 0.f, 0.f, 0.f, 0.f, 0.f, 0.f, 0.f};
    return z;
}

__device__ __forceinline__ v8f wmma_f16(v16h a, v16h b, v8f c) {
    // emits v_wmma_f32_16x16x32_f16
    return __builtin_amdgcn_wmma_f32_16x16x32_f16(false, a, false, b,
                                                  (short)0, c, false, false);
}

// Fragment load for 16-bit A/B (16x32, MxK), lane layout per CDNA5 ISA 7.12.2:
// lanes 0-15 = row M, khalf = lane>>4 selects the +8 K-half.
// elems 0..7  = K kbase+khalf*8+0..7    (VGPR 0-3)
// elems 8..15 = K kbase+16+khalf*8+0..7 (VGPR 4-7)
// p must point at row + kbase + khalf*8 (16-byte aligned).
__device__ __forceinline__ v16h frag_ld(const _Float16* p) {
    union { v16h v; v8h h[2]; } u;
    u.h[0] = *reinterpret_cast<const v8h*>(p);
    u.h[1] = *reinterpret_cast<const v8h*>(p + 16);
    return u.v;
}

// Low-half-only fragment (K 0..15 valid, K 16..31 forced to zero).
__device__ __forceinline__ v16h frag_ld_lo(const _Float16* p) {
    union { v16h v; v8h h[2]; _Float16 s[16]; } u;
    u.h[0] = *reinterpret_cast<const v8h*>(p);
#pragma unroll
    for (int e = 8; e < 16; ++e) u.s[e] = (_Float16)0.f;
    return u.v;
}

// ============================================================================
// Prep kernels: one-time (per call) f32 -> f16 conversion / transpose so the
// hot GEMM stages with pure b128 copies and half the L2 weight bytes.
// ============================================================================
__global__ __launch_bounds__(256)
void prep_xe_kernel(const float* __restrict__ x, const float* __restrict__ pos,
                    _Float16* __restrict__ XEf) {
    int idx = blockIdx.x * 256 + threadIdx.x;      // < TOK*KPAD (exact)
    int token = idx / KPAD;
    int kk    = idx - token * KPAD;
    float v = 0.f;
    if (kk < DIN)      v = x[token * DIN + kk];
    else if (kk < 70)  v = pos[(token & 15) * 6 + (kk - DIN)];
    XEf[idx] = (_Float16)v;
}

__global__ __launch_bounds__(256)
void prep_wht_kernel(const float* __restrict__ W0, const float* __restrict__ W1,
                     const float* __restrict__ W2, _Float16* __restrict__ WhT) {
    const float* W = (blockIdx.y == 0) ? W0 : (blockIdx.y == 1) ? W1 : W2;
    _Float16* dst  = WhT + (size_t)blockIdx.y * (CDIM * KPAD);
    int idx = blockIdx.x * 256 + threadIdx.x;      // < 512*96 (exact)
    int n  = idx / KPAD;
    int kk = idx - n * KPAD;
    dst[idx] = (kk < 70) ? (_Float16)W[kk * CDIM + n] : (_Float16)0.f;
}

__global__ __launch_bounds__(256)
void prep_wot_kernel(const float* __restrict__ W0, const float* __restrict__ W1,
                     const float* __restrict__ W2, _Float16* __restrict__ WoT) {
    const float* W = (blockIdx.y == 0) ? W0 : (blockIdx.y == 1) ? W1 : W2;
    _Float16* dst  = WoT + (size_t)blockIdx.y * (CDIM * CDIM);
    int idx = blockIdx.x * 256 + threadIdx.x;      // < 512*512 (exact)
    int n  = idx >> 9;
    int kk = idx & 511;
    dst[idx] = (_Float16)W[kk * CDIM + n];         // dst[n*512+kk] = W^T
}

// ============================================================================
// Kernel 1: fused 2-layer MLP (silu hidden) for K / Q / V.
//   grid = (TOK/128, 3) ; block = 512 (16 waves) ; dynamic LDS = 253952 B
//   Each wave owns one 16-row M-tile and a 256-col N-range, processed as two
//   chunks of 8 N-tiles (acc[8] = 64 VGPRs) to stay within the 4-waves/SIMD
//   VGPR budget. All inner LDS addresses are base + constant (DS imm offsets).
//   All staging is 128-bit copies of pre-converted f16 data.
// ============================================================================
__global__ __launch_bounds__(512, 1)
void mlp_kqv_kernel(const _Float16* __restrict__ XEf,
                    const _Float16* __restrict__ WhT,
                    const _Float16* __restrict__ WoT,
                    const float* __restrict__ bh0, const float* __restrict__ bh1,
                    const float* __restrict__ bh2,
                    const float* __restrict__ bo0, const float* __restrict__ bo1,
                    const float* __restrict__ bo2,
                    _Float16* __restrict__ Kf, _Float16* __restrict__ Qf,
                    _Float16* __restrict__ Vf) {
    extern __shared__ char smem[];
    _Float16* sH  = (_Float16*)(smem);                   // 128*512*2 = 131072
    _Float16* sXE = (_Float16*)(smem + 131072);          // 128*96*2  =  24576
    _Float16* sB1 = (_Float16*)(smem + 131072 + 24576);  // 512*96*2  =  98304
    _Float16* sB2 = (_Float16*)(smem + 131072);          // 256*64*2 (aliases)

    const int tid = threadIdx.x;
    const int mlp = blockIdx.y;
    const int m0  = blockIdx.x * 128;

    const _Float16* WhTm = WhT + (size_t)mlp * (CDIM * KPAD);
    const _Float16* WoTm = WoT + (size_t)mlp * (CDIM * CDIM);
    const float* bh = (mlp == 0) ? bh0 : (mlp == 1) ? bh1 : bh2;
    const float* bo = (mlp == 0) ? bo0 : (mlp == 1) ? bo1 : bo2;
    _Float16*   out = (mlp == 0) ? Kf  : (mlp == 1) ? Qf  : Vf;

    // ---- stage XE tile + Wh^T (pure b128 copies) ----
    {
        const uint4* sx = (const uint4*)(XEf + (size_t)m0 * KPAD);
        uint4*       dx = (uint4*)sXE;
        for (int ch = tid; ch < 128 * KPAD / 8; ch += 512) dx[ch] = sx[ch];
        const uint4* sw = (const uint4*)WhTm;
        uint4*       dw = (uint4*)sB1;
        for (int ch = tid; ch < CDIM * KPAD / 8; ch += 512) dw[ch] = sw[ch];
    }
    __syncthreads();

    const int lane  = tid & 31;
    const int wave  = tid >> 5;
    const int mrow  = lane & 15;   // A row / B col within tile
    const int khalf = lane >> 4;   // K-half select
    const int mtile = wave >> 1;            // 0..7  (rows mtile*16..+15)
    const int nbase = (wave & 1) << 4;      // ntile base: 0 or 16

    const _Float16* abase1 = sXE + (mtile * 16 + mrow) * KPAD + khalf * 8;

    // ---------------- layer 1: K = 96 (3 WMMA k-steps), 2 N-chunks ----------
#pragma unroll
    for (int c = 0; c < 2; ++c) {
        v8f acc[8];
#pragma unroll
        for (int t = 0; t < 8; ++t) acc[t] = vzero8f();

        const _Float16* bbase1 =
            sB1 + ((nbase + c * 8) * 16 + mrow) * KPAD + khalf * 8;
#pragma unroll
        for (int ks = 0; ks < 3; ++ks) {
            v16h a = frag_ld(abase1 + ks * 32);
#pragma unroll
            for (int nt = 0; nt < 8; ++nt) {
                acc[nt] = wmma_f16(a, frag_ld(bbase1 + nt * (16 * KPAD) + ks * 32),
                                   acc[nt]);
            }
        }
        // bias + silu -> sH (f16)
#pragma unroll
        for (int nt = 0; nt < 8; ++nt) {
            int col = ((nbase + c * 8 + nt) << 4) + mrow;
            float bias = bh[col];
#pragma unroll
            for (int r = 0; r < 8; ++r) {
                int row = mtile * 16 + khalf * 8 + r;  // C layout: M=r (+8 hi)
                float hv  = acc[nt][r] + bias;
                float act = hv / (1.f + __expf(-hv));  // silu
                sH[row * CDIM + col] = (_Float16)act;
            }
        }
    }

    // ------- layer 2: K = 512, Wo^T streamed in [256 col x 64 k] tiles ------
    const _Float16* abase2 = sH + (mtile * 16 + mrow) * CDIM + khalf * 8;

#pragma unroll
    for (int c = 0; c < 2; ++c) {
        v8f acc2[8];
#pragma unroll
        for (int t = 0; t < 8; ++t) acc2[t] = vzero8f();

        const _Float16* bbase2 = sB2 + (nbase * 8 + mrow) * 64 + khalf * 8;

        for (int kt = 0; kt < 8; ++kt) {
            __syncthreads();   // previous sB2 (or sB1/sXE/sH-producer) done
            // stage Wo^T cols for this chunk (b128 copies):
            //   l<128 -> global col c*128+l ; l>=128 -> 256+c*128+(l-128)
            for (int ch = tid; ch < 2048; ch += 512) {
                int l  = ch >> 3;        // local col 0..255
                int c8 = ch & 7;         // uint4 within 64-half row
                int n  = (l < 128) ? (c * 128 + l) : (128 + c * 128 + l);
                ((uint4*)sB2)[ch] =
                    *(const uint4*)(WoTm + (size_t)n * CDIM + kt * 64 + c8 * 8);
            }
            // prefetch next K-tile (one 128B line per needed row)
            if (kt < 7 && tid < 256) {
                int l = tid;
                int n = (l < 128) ? (c * 128 + l) : (128 + c * 128 + l);
                __builtin_prefetch(WoTm + (size_t)n * CDIM + (kt + 1) * 64, 0, 1);
            }
            __syncthreads();
#pragma unroll
            for (int ks = 0; ks < 2; ++ks) {
                v16h a = frag_ld(abase2 + kt * 64 + ks * 32);
#pragma unroll
                for (int nt = 0; nt < 8; ++nt) {
                    acc2[nt] = wmma_f16(
                        a, frag_ld(bbase2 + nt * 1024 + ks * 32), acc2[nt]);
                }
            }
        }
        // bias + store f16 K/Q/V rows
#pragma unroll
        for (int nt = 0; nt < 8; ++nt) {
            int col = ((nbase + c * 8 + nt) << 4) + mrow;
            float bias = bo[col];
#pragma unroll
            for (int r = 0; r < 8; ++r) {
                int token = m0 + mtile * 16 + khalf * 8 + r;
                out[(size_t)token * CDIM + col] = (_Float16)(acc2[nt][r] + bias);
            }
        }
    }
}

// ============================================================================
// Kernel 2: per-batch molecular attention.
//   grid = BATCH ; block = 256 (8 waves = 8 heads)
//   scores S = K_h (16x64) x Q_h^T via 2 WMMAs; softplus/mask/normalize in f32;
//   combine O_h = W (16x16, K zero-padded to 32) x V_h (16x64) via 4 WMMAs;
//   head reduction through LDS; f32 output.
// ============================================================================
__global__ __launch_bounds__(256, 1)
void attn_kernel(const _Float16* __restrict__ Kf, const _Float16* __restrict__ Qf,
                 const _Float16* __restrict__ Vf, const float* __restrict__ fm,
                 float* __restrict__ outp) {
    __shared__ _Float16 sV[NMOL * CDIM];          // 16 KB
    __shared__ float    sWf[HEADS * NMOL * NMOL]; //  8 KB
    __shared__ _Float16 sW16[HEADS * NMOL * NMOL];//  4 KB
    __shared__ float    sO[HEADS * NMOL * ADIM];  // 32 KB
    __shared__ float    sFM[NMOL];

    const int tid = threadIdx.x;
    const int b   = blockIdx.x;

    // stage V rows for this batch (contiguous, 128-bit copies)
    {
        const uint4* src = (const uint4*)(Vf + (size_t)b * NMOL * CDIM);
        uint4*       dst = (uint4*)sV;
        for (int ch = tid; ch < NMOL * CDIM / 8; ch += 256) dst[ch] = src[ch];
    }
    if (tid < NMOL) sFM[tid] = fm[tid];
    __syncthreads();

    const int lane  = tid & 31;
    const int c     = tid >> 5;    // head
    const int nidx  = lane & 15;
    const int khalf = lane >> 4;

    const _Float16* Kb = Kf + (size_t)b * NMOL * CDIM + c * ADIM;
    const _Float16* Qb = Qf + (size_t)b * NMOL * CDIM + c * ADIM;

    // ---- scores: S[i][j] = sum_a K[i][a] Q[j][a]  (A=K rows, B cols=Q rows)
    v8f s = vzero8f();
#pragma unroll
    for (int ks = 0; ks < 2; ++ks) {
        v16h a  = frag_ld(Kb + nidx * CDIM + ks * 32 + khalf * 8);
        v16h bq = frag_ld(Qb + nidx * CDIM + ks * 32 + khalf * 8);
        s = wmma_f16(a, bq, s);
    }
    // softplus + self-mask, stash f32 weights
#pragma unroll
    for (int r = 0; r < 8; ++r) {
        int i = khalf * 8 + r;   // C layout row
        int j = nidx;            // C layout col
        float arg = s[r] * 0.125f * sFM[i] * sFM[j];
        float w = fmaxf(arg, 0.f) + log1pf(__expf(-fabsf(arg))) + 1e-5f;
        if (i == j) w = 0.f;
        sWf[c * 256 + i * 16 + j] = w;
    }
    __syncthreads();

    // normalize over j (axis -2), fold in flow_mask[j], convert to f16
    if (tid < HEADS * NMOL) {
        int c2 = tid >> 4, i = tid & 15;
        float sum = 0.f;
        for (int j = 0; j < 16; ++j) sum += sWf[c2 * 256 + i * 16 + j];
        float inv = 1.f / sum;
        for (int j = 0; j < 16; ++j)
            sW16[c2 * 256 + i * 16 + j] =
                (_Float16)(sWf[c2 * 256 + i * 16 + j] * inv * sFM[j]);
    }
    __syncthreads();

    // ---- combine: O_h = W (16x16 over j, K padded) x V_h (16j x 64d)
    v16h aw = frag_ld_lo(sW16 + c * 256 + nidx * 16 + khalf * 8);
#pragma unroll
    for (int dt = 0; dt < 4; ++dt) {
        int d = dt * 16 + nidx;
        union { v16h v; _Float16 h[16]; } bu;
#pragma unroll
        for (int e = 0; e < 8; ++e)
            bu.h[e] = sV[(khalf * 8 + e) * CDIM + c * ADIM + d];
#pragma unroll
        for (int e = 8; e < 16; ++e) bu.h[e] = (_Float16)0.f;
        v8f o = wmma_f16(aw, bu.v, vzero8f());
#pragma unroll
        for (int r = 0; r < 8; ++r)
            sO[c * (NMOL * ADIM) + (khalf * 8 + r) * ADIM + d] = o[r];
    }
    __syncthreads();

    // head reduction + f32 output  (out[b,i,d] = b*1024 + i*64 + d)
    for (int idx = tid; idx < NMOL * ADIM; idx += 256) {
        float sum = 0.f;
#pragma unroll
        for (int cc = 0; cc < HEADS; ++cc) sum += sO[cc * (NMOL * ADIM) + idx];
        outp[(size_t)b * (NMOL * ADIM) + idx] = sum;
    }
}

// ============================================================================
extern "C" void kernel_launch(void* const* d_in, const int* in_sizes, int n_in,
                              void* d_out, int out_size, void* d_ws, size_t ws_size,
                              hipStream_t stream) {
    const float* x    = (const float*)d_in[0];
    const float* fm   = (const float*)d_in[1];
    const float* Wk_h = (const float*)d_in[2];
    const float* bk_h = (const float*)d_in[3];
    const float* Wk_o = (const float*)d_in[4];
    const float* bk_o = (const float*)d_in[5];
    const float* Wq_h = (const float*)d_in[6];
    const float* bq_h = (const float*)d_in[7];
    const float* Wq_o = (const float*)d_in[8];
    const float* bq_o = (const float*)d_in[9];
    const float* Wv_h = (const float*)d_in[10];
    const float* bv_h = (const float*)d_in[11];
    const float* Wv_o = (const float*)d_in[12];
    const float* bv_o = (const float*)d_in[13];
    const float* pos  = (const float*)d_in[14];

    // ---- workspace layout (f16) ----
    const size_t per = (size_t)TOK * CDIM;                 // 16.7M halves
    _Float16* Kf  = (_Float16*)d_ws;
    _Float16* Qf  = Kf + per;
    _Float16* Vf  = Qf + per;
    _Float16* XEf = Vf + per;                              // TOK * 96
    _Float16* WhT = XEf + (size_t)TOK * KPAD;              // 3 * 512*96
    _Float16* WoT = WhT + (size_t)3 * CDIM * KPAD;         // 3 * 512*512
    // total ~103.8 MB

    // ---- prep: f16 conversion / transposes (tiny) ----
    prep_xe_kernel<<<TOK * KPAD / 256, 256, 0, stream>>>(x, pos, XEf);
    prep_wht_kernel<<<dim3(CDIM * KPAD / 256, 3), 256, 0, stream>>>(
        Wk_h, Wq_h, Wv_h, WhT);
    prep_wot_kernel<<<dim3(CDIM * CDIM / 256, 3), 256, 0, stream>>>(
        Wk_o, Wq_o, Wv_o, WoT);

    // 131072 (sH) + 24576 (sXE) + 98304 (Wh^T)  — layer2 tile aliases sXE/sB1
    const size_t lds1 = 131072 + 24576 + 98304;   // 253952 B < 320 KB/WGP

    dim3 g1(TOK / 128, 3);
    mlp_kqv_kernel<<<g1, 512, lds1, stream>>>(
        XEf, WhT, WoT,
        bk_h, bq_h, bv_h,
        bk_o, bq_o, bv_o,
        Kf, Qf, Vf);

    attn_kernel<<<BATCH, 256, 0, stream>>>(Kf, Qf, Vf, fm, (float*)d_out);
}
